// ColorHistogramLoss_52733608460409
// MI455X (gfx1250) — matrix operand order repaired and checked by reading, not compiled, not run on hardware
//
#include <hip/hip_runtime.h>
#include <hip/hip_bf16.h>

typedef __attribute__((ext_vector_type(16))) _Float16 v16h;
typedef __attribute__((ext_vector_type(8)))  float    v8f;

#define BINS 64
#define NPIX (512 * 512)
#define NCH  12   // B*C = 4*3 channels per tensor

// ---------------------------------------------------------------------------
// Kernel 0: zero the global histogram workspace (2 tensors * 12 ch * 64 bins)
// ---------------------------------------------------------------------------
__global__ void ColorHistogramLoss_zero_kernel(float* __restrict__ p, int n) {
    int i = blockIdx.x * blockDim.x + threadIdx.x;
    if (i < n) p[i] = 0.0f;
}

// ---------------------------------------------------------------------------
// Kernel 1: soft histogram.
// Gaussian weight around bin k (center (k+0.5)/64, sigma = 1/64):
//   t = x*64 - 0.5, w(k) = exp(-0.5*(t-k)^2)
// Factorized: with k0=floor(t), f=t-k0 in [0,1):
//   w(k0+dk) = exp(-0.5 f^2) * exp(f)^dk * exp(-0.5 dk^2)
// -> 2 transcendentals + 1 rcp per pixel instead of 64 exps.
// Window dk in [-5, +6] (dropped mass < exp(-18), below the 1e-8 epsilon).
// Per-wave LDS histograms: 64 bins == 64 LDS banks, fp32 ds-atomics.
// ---------------------------------------------------------------------------
__global__ __launch_bounds__(256) void ColorHistogramLoss_hist_kernel(
    const float* __restrict__ pred, const float* __restrict__ tgt,
    float* __restrict__ ghist) {
    __shared__ float lh[8][BINS];
    const int tid  = threadIdx.x;
    const int wave = tid >> 5;
    const int lane = tid & 31;
    for (int b = lane; b < BINS; b += 32) lh[wave][b] = 0.0f;
    __syncthreads();

    const int ch = blockIdx.y;
    const float* __restrict__ src = (blockIdx.z == 0) ? pred : tgt;
    const float4* __restrict__ src4 = (const float4*)(src + (size_t)ch * NPIX);
    const int base4 = blockIdx.x * 1024;   // 64 blocks/channel * 1024 float4 = 256K pixels

    const float CQ[7] = {1.0f,            0.60653066f,    0.13533528f,
                         0.011108997f,    3.3546263e-4f,  3.7266532e-6f,
                         1.5229979e-8f};  // exp(-0.5*dk^2)

    #pragma unroll
    for (int t4 = 0; t4 < 4; ++t4) {
        float4 v = src4[base4 + t4 * 256 + tid];   // global_load_b128, fully coalesced
        float xs[4] = {v.x, v.y, v.z, v.w};
        #pragma unroll
        for (int i = 0; i < 4; ++i) {
            float t  = xs[i] * 64.0f - 0.5f;
            float kf = floorf(t);
            int   k0 = (int)kf;
            float f  = t - kf;                       // [0,1)
            float w0 = __expf(-0.5f * f * f);        // v_exp_f32
            float e1 = __expf(f);                    // v_exp_f32
            float r1 = __builtin_amdgcn_rcpf(e1);    // v_rcp_f32

            float p = w0;                            // ascending bins k0..k0+6
            #pragma unroll
            for (int dk = 0; dk <= 6; ++dk) {
                int k = k0 + dk;
                if ((unsigned)k < (unsigned)BINS) atomicAdd(&lh[wave][k], p * CQ[dk]);
                p *= e1;
            }
            p = w0;                                  // descending bins k0-1..k0-5
            #pragma unroll
            for (int dk = 1; dk <= 5; ++dk) {
                p *= r1;
                int k = k0 - dk;
                if ((unsigned)k < (unsigned)BINS) atomicAdd(&lh[wave][k], p * CQ[dk]);
            }
        }
    }
    __syncthreads();

    if (tid < BINS) {
        float s = 0.0f;
        #pragma unroll
        for (int w = 0; w < 8; ++w) s += lh[w][tid];
        atomicAdd(&ghist[((size_t)blockIdx.z * NCH + ch) * BINS + tid], s);
    }
}

// ---------------------------------------------------------------------------
// Kernel 2: finalize on one wave32.
// d = pred_hist/rowsum - tgt_hist/rowsum  (16x64, rows 12..15 zero-padded)
// |pred_cdf - tgt_cdf| = |cumsum(d)|; cumsum = d @ T with T[k][j] = (k<=j)
// computed as 4 N-tiles x 2 K-steps of v_wmma_f32_16x16x32_f16 (ones are
// exact in f16; d ~ 1e-3 so operand rounding is ~1e-6 in the scalar).
// ---------------------------------------------------------------------------
__global__ void ColorHistogramLoss_final_kernel(const float* __restrict__ gh,
                                                float* __restrict__ out) {
    __shared__ float dmat[16][BINS];
    const int lane = threadIdx.x;                    // exactly 32 threads
    for (int i = lane; i < 16 * BINS; i += 32) (&dmat[0][0])[i] = 0.0f;
    __syncthreads();

    for (int c = 0; c < NCH; ++c) {
        float p0 = gh[c * BINS + lane];
        float p1 = gh[c * BINS + lane + 32];
        float t0 = gh[NCH * BINS + c * BINS + lane];
        float t1 = gh[NCH * BINS + c * BINS + lane + 32];
        float ps = p0 + p1, ts = t0 + t1;
        #pragma unroll
        for (int off = 16; off > 0; off >>= 1) {
            ps += __shfl_xor(ps, off);
            ts += __shfl_xor(ts, off);
        }
        float ip = 1.0f / (ps + 1e-8f);
        float it = 1.0f / (ts + 1e-8f);
        dmat[c][lane]      = p0 * ip - t0 * it;
        dmat[c][lane + 32] = p1 * ip - t1 * it;
    }
    __syncthreads();

    // A fragments (ISA 7.12.2, 16-bit A 16x32): M = lane%16;
    // lanes 0-15: elems 0-7 -> K 0-7, elems 8-15 -> K 16-23
    // lanes16-31: elems 0-7 -> K 8-15, elems 8-15 -> K 24-31
    const int M  = lane & 15;
    const int hi = lane >> 4;
    v16h a0, a1;
    #pragma unroll
    for (int e = 0; e < 16; ++e) {
        int K = (hi << 3) + e + ((e >= 8) ? 8 : 0);
        a0[e] = (_Float16)dmat[M][K];
        a1[e] = (_Float16)dmat[M][32 + K];
    }

    float acc = 0.0f;
    for (int j0 = 0; j0 < BINS; j0 += 16) {
        // B fragment (32x16): N = lane%16, elem e -> K = 8*(lane>=16) + e + 8*(e>=8)
        v16h b0, b1;
        const int N  = lane & 15;
        const int kB = hi << 3;
        #pragma unroll
        for (int e = 0; e < 16; ++e) {
            int Klo = kB + e + ((e >= 8) ? 8 : 0);
            b0[e] = (_Float16)((Klo      <= j0 + N) ? 1.0f : 0.0f);
            b1[e] = (_Float16)((32 + Klo <= j0 + N) ? 1.0f : 0.0f);
        }
        v8f cacc = {};
        cacc = __builtin_amdgcn_wmma_f32_16x16x32_f16(
            false, a0, false, b0, (short)0, cacc, false, false);
        cacc = __builtin_amdgcn_wmma_f32_16x16x32_f16(
            false, a1, false, b1, (short)0, cacc, false, false);
        // C/D layout: reg r -> row M = r + 8*(lane>=16); keep rows < 12
        const int rowOff = hi << 3;
        #pragma unroll
        for (int r = 0; r < 8; ++r)
            if (r + rowOff < NCH) acc += fabsf(cacc[r]);
    }
    #pragma unroll
    for (int off = 16; off > 0; off >>= 1) acc += __shfl_xor(acc, off);
    if (lane == 0) out[0] = acc * (1.0f / (float)(NCH * BINS));
}

// ---------------------------------------------------------------------------
extern "C" void kernel_launch(void* const* d_in, const int* in_sizes, int n_in,
                              void* d_out, int out_size, void* d_ws, size_t ws_size,
                              hipStream_t stream) {
    (void)in_sizes; (void)n_in; (void)out_size; (void)ws_size;
    const float* pred = (const float*)d_in[0];
    const float* tgt  = (const float*)d_in[1];
    float* ghist = (float*)d_ws;          // 2 * 12 * 64 floats = 6 KB
    float* out   = (float*)d_out;

    const int nhist = 2 * NCH * BINS;
    ColorHistogramLoss_zero_kernel<<<(nhist + 255) / 256, 256, 0, stream>>>(ghist, nhist);
    ColorHistogramLoss_hist_kernel<<<dim3(64, NCH, 2), 256, 0, stream>>>(pred, tgt, ghist);
    ColorHistogramLoss_final_kernel<<<1, 32, 0, stream>>>(ghist, out);
}
